// MMG_single_35751307771924
// MI455X (gfx1250) — matrix operand order repaired and checked by reading, not compiled, not run on hardware
//
#include <hip/hip_runtime.h>
#include <hip/hip_bf16.h>
#include <math.h>

// ---------------------------------------------------------------------------
// MI455X (gfx1250) implementation of the 2-layer edge-attention GNN.
// Dense layers -> v_wmma_f32_16x16x32_bf16 (wave32):
//   - weights pre-transposed+cast to bf16 [Npad,K]
//   - B panel (128 cols x 64 k) staged into double-buffered LDS with
//     GLOBAL_LOAD_ASYNC_TO_LDS_B128 (ASYNCcnt + s_wait_asynccnt + barrier),
//     overlapping the current chunk's WMMAs.
//   - 16-tile chunk is a rotating 4-slot register pipeline pinned with
//     sched_barrier(0): each WMMA waits with dscnt=6 (3 tiles in flight)
//     instead of draining DS.
// Irregular graph work (gathers, BN stats, softmax-over-32, segment max/sum)
// is plain VALU + global atomics.
// Requires ws_size >= ~1.5 GB.
// ---------------------------------------------------------------------------

#define USE_ASYNC_LDS 1   // set 0 to fall back to load+ds_store staging

typedef __attribute__((ext_vector_type(16))) __bf16 v16bf;
typedef __attribute__((ext_vector_type(8)))  float  v8f;

namespace {
constexpr int   Nn   = 10000;
constexpr int   Ee   = 100000;
constexpr int   EHn  = Ee * 8;       // 800000 rows in per-head attention GEMMs
constexpr float EPS  = 1e-5f;
constexpr float INV_TEMP = 0.17677669529663687f;   // 1/sqrt(32)
constexpr int   KSTEP   = 64;        // staged k-chunk (all K are multiples)
constexpr int   BSTRIDE = 88;        // LDS elems per column (176B, 16B-aligned)
}

// copy one 16B piece global->LDS
__device__ __forceinline__ void stage_piece(const __bf16* g, __bf16* s) {
#if USE_ASYNC_LDS
  unsigned saddr = (unsigned)(uintptr_t)s;   // LDS aperture: offset = addr[31:0]
  asm volatile("global_load_async_to_lds_b128 %0, %1, off"
               :: "v"(saddr), "v"(g) : "memory");
#else
  *(uint4*)s = *(const uint4*)g;
#endif
}
__device__ __forceinline__ void stage_fence() {
#if USE_ASYNC_LDS
  asm volatile("s_wait_asynccnt 0x0" ::: "memory");
#endif
}

// ------------------------------- utility kernels ---------------------------

__global__ void k_fill_f32(float* p, long long n, float v) {
  long long i = blockIdx.x * 256LL + threadIdx.x;
  if (i < n) p[i] = v;
}
__global__ void k_fill_i32(int* p, long long n, int v) {
  long long i = blockIdx.x * 256LL + threadIdx.x;
  if (i < n) p[i] = v;
}
__global__ void k_copy_f32(const float* s, float* d, long long n) {
  long long i = blockIdx.x * 256LL + threadIdx.x;
  if (i < n) d[i] = s[i];
}

// W [K,Ncols] f32 row-major  ->  Wt [Npad,K] bf16 (zero padded rows)
__global__ void k_transpose_cast(const float* W, __bf16* Wt, int K, int Ncols,
                                 int Npad) {
  long long idx = blockIdx.x * 256LL + threadIdx.x;
  if (idx >= (long long)Npad * K) return;
  int n = (int)(idx / K);
  int k = (int)(idx - (long long)n * K);
  float v = (n < Ncols) ? W[(size_t)k * Ncols + n] : 0.f;
  Wt[idx] = (__bf16)v;
}

// ------------------------------- WMMA GEMM ---------------------------------
// C[M,Ncols] = act(A[M,K](bf16,lda) @ Bt[Npad,K]^T + bias)
// act: 0 none, 1 relu, 2 sigmoid.  storebf: write bf16 instead of f32.
// Requires K % 64 == 0 (all call sites satisfy this).
__global__ __launch_bounds__(256) void k_gemm(
    const __bf16* __restrict__ A, int lda,
    const __bf16* __restrict__ Bt, int K,
    const float* __restrict__ bias,
    float* Cf, __bf16* Cb, int ldc,
    int M, int Ncols, int act, int storebf) {
  __shared__ __bf16 smem[2][128 * BSTRIDE];   // 2 x 22528 B

  const int tid  = threadIdx.x;
  const int lane = tid & 31;
  const int wave = tid >> 5;
  const int half = lane >> 4;     // which K-half this lane holds
  const int l16  = lane & 15;
  const int m0 = blockIdx.x * 128 + wave * 16;
  const int n0 = blockIdx.y * 128;

  int arow = m0 + l16;
  if (arow >= M) arow = M - 1;                 // clamp; stores are guarded
  const __bf16* abase = A + (size_t)arow * lda + half * 8;

  // staging: thread t copies column (t>>1), elements (t&1)*32 .. +31
  const int scol = tid >> 1;
  const int soff = (tid & 1) * 32;
  const __bf16* gB = Bt + (size_t)(n0 + scol) * K + soff;
  __bf16* sB0 = &smem[0][0] + scol * BSTRIDE + soff;
  __bf16* sB1 = &smem[1][0] + scol * BSTRIDE + soff;

  // stage chunk 0 into buffer 0
  stage_piece(gB + 0,  sB0 + 0);
  stage_piece(gB + 8,  sB0 + 8);
  stage_piece(gB + 16, sB0 + 16);
  stage_piece(gB + 24, sB0 + 24);

  v8f acc[8];
  const v8f vz = {0.f, 0.f, 0.f, 0.f, 0.f, 0.f, 0.f, 0.f};
#pragma unroll
  for (int j = 0; j < 8; ++j) acc[j] = vz;

  // A prefetch for chunk 0 (two 32-wide k-steps)
  uint4 a0[2], a1[2];
  a0[0] = *(const uint4*)(abase + 0);
  a0[1] = *(const uint4*)(abase + 16);
  a1[0] = *(const uint4*)(abase + 32);
  a1[1] = *(const uint4*)(abase + 48);

  stage_fence();
  __syncthreads();

  int cbuf = 0;
  for (int kk = 0; kk < K; kk += KSTEP) {
    uint4 na0[2], na1[2];
    const bool more = (kk + KSTEP) < K;
    if (more) {
      // stage next chunk into the other buffer (overlaps with WMMAs below)
      const __bf16* g = gB + (kk + KSTEP);
      __bf16* s = cbuf ? sB0 : sB1;
      stage_piece(g + 0,  s + 0);
      stage_piece(g + 8,  s + 8);
      stage_piece(g + 16, s + 16);
      stage_piece(g + 24, s + 24);
      // prefetch next A fragments
      const __bf16* an = abase + kk + KSTEP;
      na0[0] = *(const uint4*)(an);
      na0[1] = *(const uint4*)(an + 16);
      na1[0] = *(const uint4*)(an + 32);
      na1[1] = *(const uint4*)(an + 48);
    }

    const __bf16* bl = &smem[cbuf][0] + l16 * BSTRIDE + half * 8;
    union BF { v16bf v; uint4 q[2]; };

    // 16 tiles: t = (k-step << 3) | n-tile.  Rotating 4-slot pipeline,
    // pinned with sched_barrier(0) so each WMMA keeps 3 tiles of ds_loads
    // in flight (s_wait_dscnt 6) instead of draining.
    BF b[4];
#pragma unroll
    for (int t = 0; t < 4; ++t) {
      const __bf16* bp = bl + (t & 7) * 16 * BSTRIDE + (t >> 3) * 32;
      b[t].q[0] = *(const uint4*)(bp);
      b[t].q[1] = *(const uint4*)(bp + 16);
    }
    __builtin_amdgcn_sched_barrier(0);
#pragma unroll
    for (int t = 0; t < 16; ++t) {
      BF a;
      a.q[0] = (t >> 3) ? a1[0] : a0[0];
      a.q[1] = (t >> 3) ? a1[1] : a0[1];
      acc[t & 7] = __builtin_amdgcn_wmma_f32_16x16x32_bf16(
          false, a.v, false, b[t & 3].v, (short)0, acc[t & 7], false, false);
      if (t + 4 < 16) {
        const int u = t + 4;
        const __bf16* bp = bl + (u & 7) * 16 * BSTRIDE + (u >> 3) * 32;
        b[t & 3].q[0] = *(const uint4*)(bp);
        b[t & 3].q[1] = *(const uint4*)(bp + 16);
      }
      __builtin_amdgcn_sched_barrier(0);
    }

    stage_fence();
    __syncthreads();
    if (more) {
      a0[0] = na0[0]; a0[1] = na0[1];
      a1[0] = na1[0]; a1[1] = na1[1];
    }
    cbuf ^= 1;
  }

#pragma unroll
  for (int j = 0; j < 8; ++j) {
    int c = n0 + j * 16 + l16;
    if (c >= Ncols) continue;
    float bs = bias ? bias[c] : 0.f;
#pragma unroll
    for (int r = 0; r < 8; ++r) {
      int mr = m0 + half * 8 + r;     // C layout: lanes 16-31 are rows +8
      if (mr >= M) continue;
      float v = acc[j][r] + bs;
      if (act == 1) v = fmaxf(v, 0.f);
      else if (act == 2) v = 1.f / (1.f + __expf(-v));
      size_t off = (size_t)mr * ldc + c;
      if (storebf) Cb[off] = (__bf16)v;
      else         Cf[off] = v;
    }
  }
}

// ---------------------------- graph / pointwise ----------------------------

// Ae[e, 0..1023] = [ x[row[e]] | ef[e] | rev_ef[e] | x[col[e]] ]  (bf16)
__global__ void k_gather_edge(const float* x, const float* ef, const int* row,
                              const int* col, const int* rev, __bf16* Ae) {
  long long idx = blockIdx.x * 256LL + threadIdx.x;
  if (idx >= (long long)Ee * 1024) return;
  int e = (int)(idx >> 10);
  int c = (int)(idx & 1023);
  float v;
  if (c < 256)      v = x[(size_t)row[e] * 256 + c];
  else if (c < 512) v = ef[(size_t)e * 256 + (c - 256)];
  else if (c < 768) {
    int r = rev[e];
    v = (r >= 0) ? ef[(size_t)r * 256 + (c - 512)] : 0.f;
  } else            v = x[(size_t)col[e] * 256 + (c - 768)];
  Ae[idx] = (__bf16)v;
}

// distance MLP hidden: dh[e,32] = df(e) @ dW1 + db1 ; accumulate BN stats
__global__ void k_dist_hidden(const float* pos, const int* row, const int* col,
                              const float* dW1, const float* db1, float* dh,
                              float* dst) {
  int e = blockIdx.x * 256 + threadIdx.x;
  if (e >= Ee) return;
  int r = row[e], c = col[e];
  float dx = pos[(size_t)r * 3 + 0] - pos[(size_t)c * 3 + 0];
  float dy = pos[(size_t)r * 3 + 1] - pos[(size_t)c * 3 + 1];
  float dz = pos[(size_t)r * 3 + 2] - pos[(size_t)c * 3 + 2];
  float dd = sqrtf(dx * dx + dy * dy + dz * dz);
#pragma unroll
  for (int j = 0; j < 32; ++j) {
    float h = db1[j] + dx * dW1[j] + dy * dW1[32 + j] + dz * dW1[64 + j] +
              dd * dW1[96 + j];
    dh[(size_t)e * 32 + j] = h;
    atomicAdd(&dst[j], h);
    atomicAdd(&dst[32 + j], h * h);
  }
}

// dm[e] = sigmoid( relu(bn(dh)) @ dW2 + db2 )
__global__ void k_dist_out(const float* dh, const float* dst, const float* g,
                           const float* b, const float* dW2, const float* db2,
                           float* dm) {
  int e = blockIdx.x * 256 + threadIdx.x;
  if (e >= Ee) return;
  float acc = db2[0];
#pragma unroll
  for (int j = 0; j < 32; ++j) {
    float m = dst[j] / (float)Ee;
    float var = dst[32 + j] / (float)Ee - m * m;
    float hv = (dh[(size_t)e * 32 + j] - m) * rsqrtf(var + EPS) * g[j] + b[j];
    acc += fmaxf(hv, 0.f) * dW2[j];
  }
  dm[e] = 1.f / (1.f + __expf(-acc));
}

// column sums / sum-of-squares of X[M,C] into stats[0..C)=sum, [C..2C)=sumsq
__global__ void k_colstats(const float* X, int M, int C, float* stats) {
  int r0 = blockIdx.x * 512;
  int r1 = min(r0 + 512, M);
  for (int c = threadIdx.x; c < C; c += blockDim.x) {
    float s = 0.f, s2 = 0.f;
    for (int r = r0; r < r1; ++r) {
      float v = X[(size_t)r * C + c];
      s += v; s2 += v * v;
    }
    atomicAdd(&stats[c], s);
    atomicAdd(&stats[C + c], s2);
  }
}

__global__ void k_bnrelu(const float* X, __bf16* Y, const float* stats,
                         const float* g, const float* b, int M, int C) {
  long long idx = blockIdx.x * 256LL + threadIdx.x;
  if (idx >= (long long)M * C) return;
  int c = (int)(idx % C);
  float m = stats[c] / (float)M;
  float var = stats[C + c] / (float)M - m * m;
  float v = (X[idx] - m) * rsqrtf(var + EPS) * g[c] + b[c];
  Y[idx] = (__bf16)fmaxf(v, 0.f);
}

// Aat[(e*8+h), c] = c<32 ? q[e, c*8+h] : k[e, (c-32)*8+h]
__global__ void k_gather_att(const __bf16* qb, const __bf16* kb, __bf16* Aat) {
  long long idx = blockIdx.x * 256LL + threadIdx.x;
  if (idx >= (long long)EHn * 64) return;
  int r = (int)(idx >> 6);
  int c = (int)(idx & 63);
  int e = r >> 3, h = r & 7;
  Aat[idx] = (c < 32) ? qb[(size_t)e * 256 + c * 8 + h]
                      : kb[(size_t)e * 256 + (c - 32) * 8 + h];
}

// per (e,h): softmax over 32 channels of att*dm/TEMP, in place
__global__ void k_att_softmax(float* att, const float* dm) {
  long long r = blockIdx.x * 256LL + threadIdx.x;
  if (r >= EHn) return;
  float s = dm[r >> 3] * INV_TEMP;
  float* p = att + r * 32;
  float t[32], mx = -1e30f;
#pragma unroll
  for (int c = 0; c < 32; ++c) { t[c] = p[c] * s; mx = fmaxf(mx, t[c]); }
  float sum = 0.f;
#pragma unroll
  for (int c = 0; c < 32; ++c) { t[c] = __expf(t[c] - mx); sum += t[c]; }
  float inv = 1.f / sum;
#pragma unroll
  for (int c = 0; c < 32; ++c) p[c] = t[c] * inv;
}

// wv[e,d] = prob[(e*8+h),c] * v[e,d]   with d = c*8+h
__global__ void k_wv(const float* prob, const float* v, float* wv) {
  long long idx = blockIdx.x * 256LL + threadIdx.x;
  if (idx >= (long long)Ee * 256) return;
  int e = (int)(idx >> 8), d = (int)(idx & 255);
  int c = d >> 3, h = d & 7;
  wv[idx] = prob[((size_t)(e * 8 + h)) * 32 + c] * v[idx];
}

__device__ __forceinline__ int enc_f32(float f) {
  int i = __float_as_int(f);
  return (i >= 0) ? i : (i ^ 0x7fffffff);   // order-preserving int encoding
}

__global__ void k_agg_atomic(const float* wv, const int* row, int* aggi) {
  long long idx = blockIdx.x * 256LL + threadIdx.x;
  if (idx >= (long long)Ee * 256) return;
  int e = (int)(idx >> 8), d = (int)(idx & 255);
  atomicMax(&aggi[(size_t)row[e] * 256 + d], enc_f32(wv[idx]));
}

__global__ void k_agg_decode(int* aggi, float* agg) {
  long long idx = blockIdx.x * 256LL + threadIdx.x;
  if (idx >= (long long)Nn * 256) return;
  int m = aggi[idx];
  float f = 0.f;
  if (m != (int)0x80000000) {
    int i = (m >= 0) ? m : (m ^ 0x7fffffff);
    f = __int_as_float(i);
    if (!__builtin_isfinite(f)) f = 0.f;
  }
  agg[idx] = f;   // in-place over the int buffer
}

__global__ void k_counts(const int* row, const int* col, float* oc, float* ic) {
  int e = blockIdx.x * 256 + threadIdx.x;
  if (e >= Ee) return;
  atomicAdd(&oc[row[e]], 1.f);
  atomicAdd(&ic[col[e]], 1.f);
}

__global__ void k_ue_scatter(const float* ue, const int* row, const int* col,
                             float* srow, float* scol) {
  long long idx = blockIdx.x * 256LL + threadIdx.x;
  if (idx >= (long long)Ee * 256) return;
  int e = (int)(idx >> 8), d = (int)(idx & 255);
  float v = ue[idx];
  atomicAdd(&srow[(size_t)row[e] * 256 + d], v);
  atomicAdd(&scol[(size_t)col[e] * 256 + d], v);
}

__global__ void k_twin(const float* srow, const float* scol, const float* oc,
                       const float* ic, __bf16* twb) {
  long long idx = blockIdx.x * 256LL + threadIdx.x;
  if (idx >= (long long)Nn * 512) return;
  int n = (int)(idx >> 9), d = (int)(idx & 511);
  float t = (d < 256) ? srow[(size_t)n * 256 + d] / fmaxf(oc[n], 1.f)
                      : scol[(size_t)n * 256 + (d - 256)] / fmaxf(ic[n], 1.f);
  twb[idx] = (__bf16)t;
}

__global__ void k_gather_node(const float* x, const float* agg, __bf16* An) {
  long long idx = blockIdx.x * 256LL + threadIdx.x;
  if (idx >= (long long)Nn * 512) return;
  int n = (int)(idx >> 9), d = (int)(idx & 511);
  float v = (d < 256) ? x[(size_t)n * 256 + d] : agg[(size_t)n * 256 + (d - 256)];
  An[idx] = (__bf16)v;
}

__global__ void k_final(const float* un, const float* ea, float* xb) {
  long long idx = blockIdx.x * 256LL + threadIdx.x;
  if (idx >= (long long)Nn * 256) return;
  xb[idx] = fmaxf(un[idx], 0.f) * ea[idx];
}

// out[e,c*8+h] = prob[(e*8+h),c]
__global__ void k_prob_out(const float* prob, float* out) {
  long long idx = blockIdx.x * 256LL + threadIdx.x;
  if (idx >= (long long)Ee * 256) return;
  int e = (int)(idx >> 8), d = (int)(idx & 255);
  int c = d >> 3, h = d & 7;
  out[idx] = prob[((size_t)(e * 8 + h)) * 32 + c];
}

// ------------------------------- host driver -------------------------------

extern "C" void kernel_launch(void* const* d_in, const int* in_sizes, int n_in,
                              void* d_out, int out_size, void* d_ws,
                              size_t ws_size, hipStream_t stream) {
  (void)in_sizes; (void)n_in; (void)out_size; (void)ws_size;

  const float* x_in  = (const float*)d_in[0];
  const float* ef_in = (const float*)d_in[1];
  const float* pos   = (const float*)d_in[2];
  const int*   eidx  = (const int*)d_in[3];
  const int*   rev   = (const int*)d_in[4];
  const int* row = eidx;
  const int* col = eidx + Ee;

  // params: 2 layers x 30 leaves, dict keys sorted (jax pytree order)
  enum { iWk = 0, iWq, iWv, iaW1, iaW2, iab1, iab2, ibk, ibq, ibv,
         idW1, idW2, idb1, idb2, idbe1, idg1, ieW1, ieW2, ieb1, ieb2,
         iebe1, ieg1, inW1, inW2, inb1, inb2, inbe1, ing1, itW, itb };
  auto P = [&](int l, int k) -> const float* {
    return (const float*)d_in[5 + l * 30 + k];
  };

  // ---- workspace carve (bump allocator, 256B aligned) ----
  char* wp = (char*)d_ws;
  auto alloc = [&](size_t bytes) -> void* {
    void* r = (void*)wp;
    wp += (bytes + 255) & ~(size_t)255;
    return r;
  };
  __bf16* Ae  = (__bf16*)alloc((size_t)Ee * 1024 * 2);
  float*  He  = (float*) alloc((size_t)Ee * 768 * 4);   // reused as wv
  __bf16* Hb  = (__bf16*)alloc((size_t)Ee * 768 * 2);
  float*  efb = (float*) alloc((size_t)Ee * 256 * 4);   // ue / next ef
  __bf16* qb  = (__bf16*)alloc((size_t)Ee * 256 * 2);
  __bf16* kb  = (__bf16*)alloc((size_t)Ee * 256 * 2);
  float*  vb  = (float*) alloc((size_t)Ee * 256 * 4);
  __bf16* Aat = (__bf16*)alloc((size_t)EHn * 64 * 2);
  __bf16* Hat = (__bf16*)alloc((size_t)EHn * 64 * 2);
  float*  att = (float*) alloc((size_t)EHn * 32 * 4);   // prob in place
  float*  dmb = (float*) alloc((size_t)Ee * 4);
  float*  dh  = (float*) alloc((size_t)Ee * 32 * 4);
  float*  agg = (float*) alloc((size_t)Nn * 256 * 4);   // int during atomics
  __bf16* An  = (__bf16*)alloc((size_t)Nn * 512 * 2);
  float*  Hnf = (float*) alloc((size_t)Nn * 512 * 4);
  __bf16* Hnb = (__bf16*)alloc((size_t)Nn * 512 * 2);
  float*  unb = (float*) alloc((size_t)Nn * 256 * 4);
  float*  srow= (float*) alloc((size_t)Nn * 256 * 4);
  float*  scol= (float*) alloc((size_t)Nn * 256 * 4);
  __bf16* twb = (__bf16*)alloc((size_t)Nn * 512 * 2);
  float*  eab = (float*) alloc((size_t)Nn * 256 * 4);
  float*  xb  = (float*) alloc((size_t)Nn * 256 * 4);
  float*  oc  = (float*) alloc((size_t)Nn * 4);
  float*  ic  = (float*) alloc((size_t)Nn * 4);
  float*  stats = (float*)alloc(2 * 768 * 4);
  float*  dst = (float*) alloc(64 * 4);

  __bf16 *WqT[2], *WkT[2], *WvT[2], *eW1T[2], *eW2T[2], *aW1T[2], *aW2T[2],
         *nW1T[2], *nW2T[2], *tWT[2];
  for (int l = 0; l < 2; ++l) {
    WqT[l]  = (__bf16*)alloc((size_t)256 * 256 * 2);
    WkT[l]  = (__bf16*)alloc((size_t)256 * 256 * 2);
    WvT[l]  = (__bf16*)alloc((size_t)256 * 256 * 2);
    eW1T[l] = (__bf16*)alloc((size_t)768 * 1024 * 2);
    eW2T[l] = (__bf16*)alloc((size_t)256 * 768 * 2);
    aW1T[l] = (__bf16*)alloc((size_t)128 * 64 * 2);    // padded to 128 rows
    aW2T[l] = (__bf16*)alloc((size_t)128 * 64 * 2);
    nW1T[l] = (__bf16*)alloc((size_t)512 * 512 * 2);
    nW2T[l] = (__bf16*)alloc((size_t)256 * 512 * 2);
    tWT[l]  = (__bf16*)alloc((size_t)256 * 512 * 2);
  }

  auto nb = [](long long n) { return (unsigned)((n + 255) / 256); };
  auto fillf = [&](float* p, long long n, float v) {
    k_fill_f32<<<nb(n), 256, 0, stream>>>(p, n, v);
  };
  auto tpose = [&](const float* W, __bf16* Wt, int K, int Nc) {
    int Npad = ((Nc + 127) / 128) * 128;
    k_transpose_cast<<<nb((long long)Npad * K), 256, 0, stream>>>(W, Wt, K, Nc,
                                                                  Npad);
  };
  auto gemm = [&](const __bf16* A, int lda, const __bf16* Bt, int K,
                  const float* bias, void* C, int ldc, int M, int Nc, int act,
                  int storebf) {
    dim3 g((M + 127) / 128, (Nc + 127) / 128);
    k_gemm<<<g, 256, 0, stream>>>(A, lda, Bt, K, bias, (float*)C, (__bf16*)C,
                                  ldc, M, Nc, act, storebf);
  };

  // ---- weight transposition (both layers) ----
  for (int l = 0; l < 2; ++l) {
    tpose(P(l, iWq), WqT[l], 256, 256);
    tpose(P(l, iWk), WkT[l], 256, 256);
    tpose(P(l, iWv), WvT[l], 256, 256);
    tpose(P(l, ieW1), eW1T[l], 1024, 768);
    tpose(P(l, ieW2), eW2T[l], 768, 256);
    tpose(P(l, iaW1), aW1T[l], 64, 64);
    tpose(P(l, iaW2), aW2T[l], 64, 32);
    tpose(P(l, inW1), nW1T[l], 512, 512);
    tpose(P(l, inW2), nW2T[l], 512, 256);
    tpose(P(l, itW), tWT[l], 512, 256);
  }

  // ---- degree counts (constant across layers) ----
  fillf(oc, Nn, 0.f);
  fillf(ic, Nn, 0.f);
  k_counts<<<nb(Ee), 256, 0, stream>>>(row, col, oc, ic);

  const float* xcur = x_in;
  const float* efcur = ef_in;
  for (int l = 0; l < 2; ++l) {
    // distance mask
    fillf(dst, 64, 0.f);
    k_dist_hidden<<<nb(Ee), 256, 0, stream>>>(pos, row, col, P(l, idW1),
                                              P(l, idb1), dh, dst);
    k_dist_out<<<nb(Ee), 256, 0, stream>>>(dh, dst, P(l, idg1), P(l, idbe1),
                                           P(l, idW2), P(l, idb2), dmb);
    // edge concat -> bf16
    k_gather_edge<<<nb((long long)Ee * 1024), 256, 0, stream>>>(xcur, efcur,
                                                                row, col, rev,
                                                                Ae);
    // edge update MLP: He = Ae @ eW1 + eb1 ; BN ; relu ; ue = Hb @ eW2 + eb2
    gemm(Ae, 1024, eW1T[l], 1024, P(l, ieb1), He, 768, Ee, 768, 0, 0);
    fillf(stats, 2 * 768, 0.f);
    k_colstats<<<(Ee + 511) / 512, 256, 0, stream>>>(He, Ee, 768, stats);
    k_bnrelu<<<nb((long long)Ee * 768), 256, 0, stream>>>(
        He, Hb, stats, P(l, ieg1), P(l, iebe1), Ee, 768);
    gemm(Hb, 768, eW2T[l], 768, P(l, ieb2), efb, 256, Ee, 256, 0, 0);
    // q/k/v from slices of Ae (lda = 1024)
    gemm(Ae, 1024, WqT[l], 256, P(l, ibq), qb, 256, Ee, 256, 0, 1);
    gemm(Ae + 256, 1024, WkT[l], 256, P(l, ibk), kb, 256, Ee, 256, 0, 1);
    gemm(Ae + 768, 1024, WvT[l], 256, P(l, ibv), vb, 256, Ee, 256, 0, 0);
    // per-head attention MLP
    k_gather_att<<<nb((long long)EHn * 64), 256, 0, stream>>>(qb, kb, Aat);
    gemm(Aat, 64, aW1T[l], 64, P(l, iab1), Hat, 64, EHn, 64, 1, 1);
    gemm(Hat, 64, aW2T[l], 64, P(l, iab2), att, 32, EHn, 32, 0, 0);
    k_att_softmax<<<nb(EHn), 256, 0, stream>>>(att, dmb);
    float* wv = He;  // reuse He region
    k_wv<<<nb((long long)Ee * 256), 256, 0, stream>>>(att, vb, wv);
    // segment max by row
    k_fill_i32<<<nb((long long)Nn * 256), 256, 0, stream>>>(
        (int*)agg, (long long)Nn * 256, (int)0x80000000);
    k_agg_atomic<<<nb((long long)Ee * 256), 256, 0, stream>>>(wv, row,
                                                              (int*)agg);
    k_agg_decode<<<nb((long long)Nn * 256), 256, 0, stream>>>((int*)agg, agg);
    // node update MLP
    k_gather_node<<<nb((long long)Nn * 512), 256, 0, stream>>>(xcur, agg, An);
    gemm(An, 512, nW1T[l], 512, P(l, inb1), Hnf, 512, Nn, 512, 0, 0);
    fillf(stats, 2 * 512, 0.f);
    k_colstats<<<(Nn + 511) / 512, 256, 0, stream>>>(Hnf, Nn, 512, stats);
    k_bnrelu<<<nb((long long)Nn * 512), 256, 0, stream>>>(
        Hnf, Hnb, stats, P(l, ing1), P(l, inbe1), Nn, 512);
    gemm(Hnb, 512, nW2T[l], 512, P(l, inb2), unb, 256, Nn, 256, 0, 0);
    // twin edge aggregation + sigmoid gate
    fillf(srow, (long long)Nn * 256, 0.f);
    fillf(scol, (long long)Nn * 256, 0.f);
    k_ue_scatter<<<nb((long long)Ee * 256), 256, 0, stream>>>(efb, row, col,
                                                              srow, scol);
    k_twin<<<nb((long long)Nn * 512), 256, 0, stream>>>(srow, scol, oc, ic,
                                                        twb);
    gemm(twb, 512, tWT[l], 512, P(l, itb), eab, 256, Nn, 256, 2, 0);
    k_final<<<nb((long long)Nn * 256), 256, 0, stream>>>(unb, eab, xb);
    xcur = xb;
    efcur = efb;
  }

  // ---- outputs: x [N,256] | ef [E,256] | prob [E,32,8] ----
  float* out_x  = (float*)d_out;
  float* out_ef = out_x + (size_t)Nn * 256;
  float* out_pr = out_ef + (size_t)Ee * 256;
  k_copy_f32<<<nb((long long)Nn * 256), 256, 0, stream>>>(xb, out_x,
                                                          (long long)Nn * 256);
  k_copy_f32<<<nb((long long)Ee * 256), 256, 0, stream>>>(efb, out_ef,
                                                          (long long)Ee * 256);
  k_prob_out<<<nb((long long)Ee * 256), 256, 0, stream>>>(att, out_pr);
}
